// MyModel_61933428416272
// MI455X (gfx1250) — matrix operand order repaired and checked by reading, not compiled, not run on hardware
//
#include <hip/hip_runtime.h>

typedef __attribute__((ext_vector_type(16))) _Float16 v16h;
typedef __attribute__((ext_vector_type(8)))  float    v8f;
typedef __attribute__((ext_vector_type(4)))  float    v4f;

constexpr int B = 8;
constexpr int S = 4096;
constexpr int E = 256;

// workspace layout (floats). All matrix regions padded to 16 rows for WMMA.
constexpr int WS_SX   = 0;          // [16,256] masked column sums of x
constexpr int WS_SQ   = 16 * E;     // [16,256] s_q = s_x @ Wq^T + L*bq
constexpr int WS_U    = 32 * E;     // [16,256] u = s_q @ Wk
constexpr int WS_T    = 48 * E;     // [16,256] t = sum alpha_s * x_s
constexpr int WS_A    = 64 * E;     // [16]     A_b = sum alpha_s
constexpr int WS_N    = 64 * E + 16;

// ---------------------------------------------------------------------------
__global__ __launch_bounds__(256) void zero_ws_kernel(float* ws, int n) {
    int i = blockIdx.x * blockDim.x + threadIdx.x;
    if (i < n) ws[i] = 0.0f;
}

// ---------------------------------------------------------------------------
// s_x[b,e] = sum_{s < L_b} x[b,s,e]
// grid: (64 chunks, B), 256 threads. Thread (rg=tid>>6, tq=tid&63) streams
// rows s ≡ rg (mod 4) of its 64-row chunk, loading float4 at e = 4*tq (b128).
__global__ __launch_bounds__(256) void colsum_kernel(const float* __restrict__ x,
                                                     const int*   __restrict__ lens,
                                                     float* __restrict__ s_x) {
    __shared__ v4f sh[256];
    const int b  = blockIdx.y;
    const int tq = threadIdx.x & 63;
    const int rg = threadIdx.x >> 6;
    const int L  = lens[b];
    const int s0 = blockIdx.x * (S / 64);
    int s1 = s0 + (S / 64);
    if (s1 > L) s1 = L;

    v4f acc = {};
    const v4f* xp = (const v4f*)(x + ((size_t)b * S + s0 + rg) * E) + tq;
    for (int s = s0 + rg; s < s1; s += 4, xp += E) acc += *xp;   // stride 4 rows

    sh[threadIdx.x] = acc;
    __syncthreads();
    if (threadIdx.x < 64) {
        v4f tot = sh[threadIdx.x] + sh[threadIdx.x + 64] +
                  sh[threadIdx.x + 128] + sh[threadIdx.x + 192];
        #pragma unroll
        for (int c = 0; c < 4; ++c)
            atomicAdd(&s_x[b * E + 4 * threadIdx.x + c], tot[c]);
    }
}

// ---------------------------------------------------------------------------
// Y[16,256] = f16(A[16,256]) @ f16(op(W)) (+ rowscale ⊗ bias), all * outScale
//   TRANSPOSE_W: B[k,n] = W[n,k] (torch Linear);   else B[k,n] = W[k,n]
//   RS_MODE: 0 = no bias; 1 = rowscale from lens (L_b); 2 = rowscale from rsf
// One workgroup, 8 waves; wave w owns N columns [32w,32w+32) as two 16x16
// tiles with a shared A fragment per K32 step. 16 static WMMAs per wave.
template <bool TRANSPOSE_W, int RS_MODE>
__global__ __launch_bounds__(256) void gemm16_wmma_kernel(
        const float* __restrict__ A,     // [16,256] (rows 8..15 zero)
        const float* __restrict__ W,     // [256,256]
        const float* __restrict__ bias,  // [256]  (RS_MODE != 0)
        const int*   __restrict__ lens,  // [8]    (RS_MODE == 1)
        const float* __restrict__ rsf,   // [16]   (RS_MODE == 2)
        float outScale,
        float* __restrict__ Y, int outRowLimit) {
    const int lane = threadIdx.x & 31;
    const int wave = threadIdx.x >> 5;
    const int khi  = lane >> 4;      // 0: lanes 0-15, 1: lanes 16-31
    const int ml   = lane & 15;
    const int n0   = wave * 32 + ml; // first owned column
    const int n1   = n0 + 16;        // second owned column

    float rsv[8];
    #pragma unroll
    for (int r = 0; r < 8; ++r) {
        const int M = r + khi * 8;
        if (RS_MODE == 1)      rsv[r] = (M < B) ? (float)lens[M] : 0.0f;
        else if (RS_MODE == 2) rsv[r] = rsf[M];
        else                   rsv[r] = 0.0f;
    }

    v8f acc0 = {}, acc1 = {};
    #pragma unroll
    for (int kk = 0; kk < 8; ++kk) {
        const int kbase = kk * 32;
        // A fragment: 16-bit A 16x32, contiguous k order (matches B)
        //   lanes 0-15 : halves 0..15 = K+0..15
        //   lanes 16-31: halves 0..15 = K+16..31
        v16h a;
        const float* arow = A + ml * E + kbase + khi * 16;
        #pragma unroll
        for (int j = 0; j < 8; ++j) a[j]     = (_Float16)arow[j];
        #pragma unroll
        for (int j = 0; j < 8; ++j) a[8 + j] = (_Float16)arow[8 + j];
        // B fragments: 16-bit B 32x16: lane group 0-15 holds K+0..15 of its
        // column, lane group 16-31 holds K+16..31.
        const int k0 = kbase + khi * 16;
        v16h b0, b1;
        if (TRANSPOSE_W) {
            const float* w0 = W + n0 * E + k0;
            const float* w1 = W + n1 * E + k0;
            #pragma unroll
            for (int j = 0; j < 16; ++j) b0[j] = (_Float16)w0[j];
            #pragma unroll
            for (int j = 0; j < 16; ++j) b1[j] = (_Float16)w1[j];
        } else {
            const float* wc = W + k0 * E;
            #pragma unroll
            for (int j = 0; j < 16; ++j) b0[j] = (_Float16)wc[j * E + n0];
            #pragma unroll
            for (int j = 0; j < 16; ++j) b1[j] = (_Float16)wc[j * E + n1];
        }
        acc0 = __builtin_amdgcn_wmma_f32_16x16x32_f16(
                   false, a, false, b0, (short)0, acc0, false, false);
        acc1 = __builtin_amdgcn_wmma_f32_16x16x32_f16(
                   false, a, false, b1, (short)0, acc1, false, false);
    }

    // C/D layout: VGPR r, lane l -> M = r + (l>=16 ? 8 : 0), N = base + (l&15)
    float bn0 = 0.0f, bn1 = 0.0f;
    if (RS_MODE != 0) { bn0 = bias[n0]; bn1 = bias[n1]; }
    #pragma unroll
    for (int r = 0; r < 8; ++r) {
        const int M = r + khi * 8;
        float y0 = acc0[r];
        float y1 = acc1[r];
        if (RS_MODE != 0) { y0 += rsv[r] * bn0; y1 += rsv[r] * bn1; }
        y0 *= outScale;
        y1 *= outScale;
        if (M < outRowLimit) {
            Y[M * E + n0] = y0;
            Y[M * E + n1] = y1;
        }
    }
}

// ---------------------------------------------------------------------------
// Pass 2: alpha_s = x_s . u[b] + (s_q[b] . bk);  t[b] += alpha_s*x_s; A[b] += alpha_s
// grid: (64 chunks, B), 256 threads; wave w owns rows s ≡ w (mod 8) of its
// 64-row chunk; lane l owns e in {4l..4l+3} ∪ {128+4l..128+4l+3} (b128 loads).
__global__ __launch_bounds__(256) void alpha_accum_kernel(
        const float* __restrict__ x,
        const int*   __restrict__ lens,
        const float* __restrict__ u,    // ws+WS_U
        const float* __restrict__ s_q,  // ws+WS_SQ
        const float* __restrict__ bk,   // [256]
        float* __restrict__ t,          // ws+WS_T
        float* __restrict__ Asum) {     // ws+WS_A
    __shared__ float red[256];
    __shared__ float sacc[8][E];

    const int b    = blockIdx.y;
    const int tid  = threadIdx.x;
    const int lane = tid & 31;
    const int wave = tid >> 5;

    // cbias = s_q[b] . bk  (block-wide reduction)
    red[tid] = s_q[b * E + tid] * bk[tid];
    __syncthreads();
    for (int off = 128; off > 0; off >>= 1) {
        if (tid < off) red[tid] += red[tid + off];
        __syncthreads();
    }
    const float cbias = red[0];

    // per-lane slice of u is loop-invariant -> registers, no LDS
    const float* ub = u + b * E + 4 * lane;
    v4f u0 = *(const v4f*)ub;
    v4f u1 = *(const v4f*)(ub + 128);

    const int L  = lens[b];
    const int s0 = blockIdx.x * (S / 64);
    int s1 = s0 + (S / 64);
    if (s1 > L) s1 = L;

    v4f t0 = {}, t1 = {};
    float asum = 0.0f;

    for (int s = s0 + wave; s < s1; s += 8) {
        const float* xrow = x + ((size_t)b * S + s) * E + 4 * lane;
        if (s + 8 < s1) __builtin_prefetch(xrow + 8 * E, 0, 0);
        const v4f x0 = *(const v4f*)xrow;           // e = 4l .. 4l+3
        const v4f x1 = *(const v4f*)(xrow + 128);   // e = 128+4l ..
        float dp = x0[0]*u0[0] + x0[1]*u0[1] + x0[2]*u0[2] + x0[3]*u0[3]
                 + x1[0]*u1[0] + x1[1]*u1[1] + x1[2]*u1[2] + x1[3]*u1[3];
        #pragma unroll
        for (int off = 16; off > 0; off >>= 1)
            dp += __shfl_xor(dp, off, 32);          // full dot in all lanes
        const float alpha = dp + cbias;
        asum += alpha;
        t0 += alpha * x0;
        t1 += alpha * x1;
    }

    __syncthreads();
    #pragma unroll
    for (int c = 0; c < 4; ++c) {
        sacc[wave][4 * lane + c]       = t0[c];
        sacc[wave][128 + 4 * lane + c] = t1[c];
    }
    __syncthreads();
    float tot = 0.0f;
    #pragma unroll
    for (int w = 0; w < 8; ++w) tot += sacc[w][tid];
    atomicAdd(&t[b * E + tid], tot);
    if (lane == 0) atomicAdd(&Asum[b], asum);       // alpha uniform per wave
}

// ---------------------------------------------------------------------------
extern "C" void kernel_launch(void* const* d_in, const int* in_sizes, int n_in,
                              void* d_out, int out_size, void* d_ws, size_t ws_size,
                              hipStream_t stream) {
    const float* x    = (const float*)d_in[0];
    const float* Wq   = (const float*)d_in[1];
    const float* bq   = (const float*)d_in[2];
    const float* Wk   = (const float*)d_in[3];
    const float* bk   = (const float*)d_in[4];
    const float* Wv   = (const float*)d_in[5];
    const float* bv   = (const float*)d_in[6];
    const int*   lens = (const int*)d_in[7];
    float* out = (float*)d_out;
    float* ws  = (float*)d_ws;

    float* s_x  = ws + WS_SX;
    float* s_q  = ws + WS_SQ;
    float* u    = ws + WS_U;
    float* t    = ws + WS_T;
    float* Asum = ws + WS_A;

    // 1) zero scratch (WMMA pad rows must be exact zeros)
    zero_ws_kernel<<<(WS_N + 255) / 256, 256, 0, stream>>>(ws, WS_N);
    // 2) masked column sums of x -> s_x
    colsum_kernel<<<dim3(64, B), 256, 0, stream>>>(x, lens, s_x);
    // 3) s_q = s_x @ Wq^T + L*bq
    gemm16_wmma_kernel<true, 1><<<1, 256, 0, stream>>>(
        s_x, Wq, bq, lens, nullptr, 1.0f, s_q, 16);
    // 4) u = s_q @ Wk   (row-wise Wk^T s_q)
    gemm16_wmma_kernel<false, 0><<<1, 256, 0, stream>>>(
        s_q, Wk, nullptr, nullptr, nullptr, 1.0f, u, 16);
    // 5) alpha pass over x -> t, Asum
    alpha_accum_kernel<<<dim3(64, B), 256, 0, stream>>>(
        x, lens, u, s_q, bk, t, Asum);
    // 6) out = (t @ Wv^T + Asum*bv) / S
    gemm16_wmma_kernel<true, 2><<<1, 256, 0, stream>>>(
        t, Wv, bv, nullptr, Asum, 1.0f / (float)S, out, B);
}
